// PureTorchMamba_68942815035945
// MI455X (gfx1250) — matrix operand order repaired and checked by reading, not compiled, not run on hardware
//
#include <hip/hip_runtime.h>
#include <hip/hip_bf16.h>

#define D_MODEL 1024
#define D_STATE 16
#define D_CONV 4
#define D_INNER 2048
#define B_SZ 2
#define L_SEQ 2048
#define BL (B_SZ * L_SEQ)              /* 4096 rows of every GEMM */
#define N_XZ (2 * D_INNER)             /* 4096 */
#define N_DBC (2 * D_STATE + D_INNER)  /* 2080 */

typedef __attribute__((ext_vector_type(16))) __bf16   v16bf;
typedef __attribute__((ext_vector_type(8)))  float    v8f;
typedef __attribute__((ext_vector_type(8)))  float    f32x8;
typedef __attribute__((ext_vector_type(8)))  __bf16   bf16x8;
typedef __attribute__((ext_vector_type(4)))  __bf16   bf16x4;
typedef __attribute__((ext_vector_type(4)))  float    f32x4;
typedef __attribute__((ext_vector_type(4)))  unsigned u32x4;
typedef __attribute__((ext_vector_type(2)))  unsigned u32x2;
typedef __attribute__((ext_vector_type(8)))  int      i32x8;
typedef __attribute__((ext_vector_type(4)))  int      i32x4;

struct BfPair { u32x4 lo, hi; };   // 32 bytes == sizeof(v16bf)

#if defined(__has_builtin)
#  if __has_builtin(__builtin_amdgcn_tensor_load_to_lds)
#    define HAVE_TDM 1
#  endif
#endif
#ifndef HAVE_TDM
#  define HAVE_TDM 0
#endif

// Convert 16 fp32 (4x f32x4) -> 16 bf16, two 16B LDS writes (native v_cvt_pk_bf16).
__device__ __forceinline__ void pack16(const f32x4* f, unsigned short* dst) {
  f32x8 a, b;
#pragma unroll
  for (int i = 0; i < 4; ++i) {
    a[i] = f[0][i]; a[4 + i] = f[1][i];
    b[i] = f[2][i]; b[4 + i] = f[3][i];
  }
  bf16x8 ha = __builtin_convertvector(a, bf16x8);
  bf16x8 hb = __builtin_convertvector(b, bf16x8);
  ((u32x4*)dst)[0] = __builtin_bit_cast(u32x4, ha);
  ((u32x4*)dst)[1] = __builtin_bit_cast(u32x4, hb);
}

#if HAVE_TDM
// Issue one TDM descriptor: copy a (tile_dim1=128 rows) x (tile_dim0=64 bf16)
// tile from a row-major bf16 matrix (row stride ldb elements) into LDS at
// lds_off, inserting 16B of pad after every 128B row -> 144B LDS row stride.
// OOB rows (>= remN) read as zero (free ragged-N handling).
__device__ __forceinline__ void tdm_load_b_tile(const unsigned short* gsrc,
                                                unsigned lds_off, int remK,
                                                int remN, int ldb) {
  const unsigned long long ga = (unsigned long long)(size_t)gsrc;
  u32x4 g0;
  g0[0] = 1u;                                        // count=1 (valid user D#)
  g0[1] = lds_off;                                   // lds_addr (bytes)
  g0[2] = (unsigned)(ga & 0xFFFFFFFFull);            // global_addr[31:0]
  g0[3] = (unsigned)((ga >> 32) & 0x01FFFFFFull)     // global_addr[56:32]
        | (2u << 30);                                // type=2 ("image")
  i32x8 g1;
  g1[0] = (1 << 16)      /* data_size=2B */
        | (1 << 20)      /* pad_enable */
        | (4 << 22)      /* pad_interval: 32 DWORDs (=128B row) */
        | (3 << 25);     /* pad_amount: 4 DWORDs (=16B) */
  g1[1] = (int)(((unsigned)remK & 0xFFFFu) << 16);   // tensor_dim0[15:0]
  g1[2] = (int)((((unsigned)remK >> 16) & 0xFFFFu)   // tensor_dim0[31:16]
        | (((unsigned)remN & 0xFFFFu) << 16));       // tensor_dim1[15:0]
  g1[3] = (int)((((unsigned)remN >> 16) & 0xFFFFu)   // tensor_dim1[31:16]
        | (64u << 16));                              // tile_dim0 = 64
  g1[4] = 128;                                       // tile_dim1=128, tile_dim2=0
  g1[5] = ldb;                                       // tensor_dim0_stride[31:0]
  g1[6] = 0;
  g1[7] = 0;
  i32x4 gz;  gz[0] = gz[1] = gz[2] = gz[3] = 0;
#if __clang_major__ >= 23
  i32x8 gz8;
#pragma unroll
  for (int i = 0; i < 8; ++i) gz8[i] = 0;
  __builtin_amdgcn_tensor_load_to_lds(g0, g1, gz, gz, gz8, 0);
#else
  __builtin_amdgcn_tensor_load_to_lds(g0, g1, gz, gz, 0);
#endif
}
#endif

// ---------------------------------------------------------------------------
// C[m,n] = sum_k A[m,k] * Bbf[n,k]  (A fp32, B pre-converted bf16)
// Block tile 128x128, 256 threads (8 waves), wave tile 64x32 (4x2 WMMA tiles).
// K stepped by 64; double-buffered LDS. A: global fp32 -> regs -> cvt -> LDS
// (register-pipelined). B: Tensor Data Mover DMA straight into padded LDS
// rows, double-buffered on TENSORcnt (manual bf16 copy if no TDM builtin).
// ---------------------------------------------------------------------------
__global__ void __launch_bounds__(256) mamba_gemm_wmma_bf16(
    const float* __restrict__ A, const unsigned short* __restrict__ Bbf,
    float* __restrict__ C, int N, int K, int lda, int ldb, int ldc) {
  constexpr int LD = 72;                     // 64 + 8 pad halfwords (144B rows)
  __shared__ unsigned short sA[2][128 * LD];
  __shared__ unsigned short sB[2][128 * LD];

  const int tid   = threadIdx.x;
  const int lane  = tid & 31;
  const int wave  = tid >> 5;
  const int wm    = wave & 1;                // 2 waves along M
  const int wn    = wave >> 1;               // 4 waves along N
  const int laneN = lane & 15;
  const int khalf = lane >> 4;

  const int m0 = blockIdx.x * 128;
  const int n0 = blockIdx.y * 128;

  const int srow  = tid >> 1;                // 0..127
  const int shalf = tid & 1;

  v8f acc[4][2];
#pragma unroll
  for (int i = 0; i < 4; ++i)
#pragma unroll
    for (int j = 0; j < 2; ++j)
#pragma unroll
      for (int e = 0; e < 8; ++e) acc[i][j][e] = 0.0f;

  const float* pa_base = A + (size_t)(m0 + srow) * lda + shalf * 16;
  unsigned short* sa_dst0 = &sA[0][srow * LD + shalf * 16];
  unsigned short* sa_dst1 = &sA[1][srow * LD + shalf * 16];

#if !HAVE_TDM
  const int nclampF = (n0 + srow < N) ? (n0 + srow) : (N - 1);
  const unsigned short* pbF = Bbf + (size_t)nclampF * ldb + shalf * 32;
#endif

  const int nslab = K >> 6;                  // K / 64 (all K multiples of 64)

#if HAVE_TDM
  if (tid == 0)                              // TDM ignores EXEC; one wave issues
    tdm_load_b_tile(Bbf + (size_t)n0 * ldb, (unsigned)(size_t)&sB[0][0],
                    K, N - n0, ldb);
#endif

  f32x4 fa[8];                               // register pipeline for A slab
#pragma unroll
  for (int i = 0; i < 4; ++i) {
    fa[i]     = *(const f32x4*)(pa_base + 4 * i);
    fa[4 + i] = *(const f32x4*)(pa_base + 32 + 4 * i);
  }

  for (int s = 0; s < nslab; ++s) {
    const int p = s & 1;
    __syncthreads();                         // compute(s-1)/(s-2) done with bufs

    pack16(&fa[0], p ? sa_dst1 : sa_dst0);
    pack16(&fa[4], (p ? sa_dst1 : sa_dst0) + 32);

#if HAVE_TDM
    if (tid == 0) {
      if (s + 1 < nslab) {
        tdm_load_b_tile(Bbf + (size_t)n0 * ldb + (s + 1) * 64,
                        (unsigned)(size_t)&sB[p ^ 1][0],
                        K - (s + 1) * 64, N - n0, ldb);
        __builtin_amdgcn_s_wait_tensorcnt((short)1);   // slab s landed
      } else {
        __builtin_amdgcn_s_wait_tensorcnt((short)0);
      }
    }
#else
    {
      const unsigned short* pb = pbF + s * 64;
      u32x4 q0 = ((const u32x4*)pb)[0], q1 = ((const u32x4*)pb)[1],
            q2 = ((const u32x4*)pb)[2], q3 = ((const u32x4*)pb)[3];
      unsigned short* d = &sB[p][srow * LD + shalf * 32];
      ((u32x4*)d)[0] = q0; ((u32x4*)d)[1] = q1;
      ((u32x4*)d)[2] = q2; ((u32x4*)d)[3] = q3;
    }
#endif

    if (s + 1 < nslab) {                     // pipeline next A slab into regs
      const float* pa = pa_base + (s + 1) * 64;
#pragma unroll
      for (int i = 0; i < 4; ++i) {
        fa[i]     = *(const f32x4*)(pa + 4 * i);
        fa[4 + i] = *(const f32x4*)(pa + 32 + 4 * i);
      }
    }

    __syncthreads();                         // sA stores + TDM-landed sB visible

#pragma unroll
    for (int ksub = 0; ksub < 2; ++ksub) {
      // A fragment (16-bit 16x32 layout): lane holds row M=laneN,
      // K in {khalf*8..+7} (VGPR0-3) and {16+khalf*8..+7} (VGPR4-7).
      v16bf af[4], bfr[2];
#pragma unroll
      for (int sm = 0; sm < 4; ++sm) {
        const unsigned short* pp =
            &sA[p][(wm * 64 + sm * 16 + laneN) * LD + ksub * 32];
        BfPair pr;
        pr.lo = *(const u32x4*)(pp + khalf * 8);
        pr.hi = *(const u32x4*)(pp + 16 + khalf * 8);
        af[sm] = __builtin_bit_cast(v16bf, pr);
      }
      // B fragment (32x16): lane holds col N=laneN, K in {khalf*8..+7} and {16+khalf*8..+7}.
#pragma unroll
      for (int sn = 0; sn < 2; ++sn) {
        const unsigned short* pp =
            &sB[p][(wn * 32 + sn * 16 + laneN) * LD + ksub * 32];
        BfPair pr;
        pr.lo = *(const u32x4*)(pp + khalf * 8);
        pr.hi = *(const u32x4*)(pp + 16 + khalf * 8);
        bfr[sn] = __builtin_bit_cast(v16bf, pr);
      }
#pragma unroll
      for (int sm = 0; sm < 4; ++sm)
#pragma unroll
        for (int sn = 0; sn < 2; ++sn)
          acc[sm][sn] = __builtin_amdgcn_wmma_f32_16x16x32_bf16(
              false, af[sm], false, bfr[sn], (short)0, acc[sm][sn], false, false);
    }
  }

  // C/D layout: VGPR r -> row (khalf*8 + r), col laneN.
#pragma unroll
  for (int sn = 0; sn < 2; ++sn) {
    const int col = n0 + wn * 32 + sn * 16 + laneN;
    if (col >= N) continue;
#pragma unroll
    for (int sm = 0; sm < 4; ++sm) {
      const int rbase = m0 + wm * 64 + sm * 16 + khalf * 8;
#pragma unroll
      for (int r = 0; r < 8; ++r)
        C[(size_t)(rbase + r) * ldc + col] = acc[sm][sn][r];
    }
  }
}

// ---------------------------------------------------------------------------
// fp32 -> bf16 weight pre-conversion (4 elements/thread).
// ---------------------------------------------------------------------------
__global__ void __launch_bounds__(256) mamba_cvt_bf16(
    const float* __restrict__ s, unsigned short* __restrict__ d, int n4) {
  const int i = blockIdx.x * 256 + threadIdx.x;
  if (i >= n4) return;
  f32x4 v = ((const f32x4*)s)[i];
  bf16x4 h = __builtin_convertvector(v, bf16x4);
  ((u32x2*)d)[i] = __builtin_bit_cast(u32x2, h);
}

// ---------------------------------------------------------------------------
// Depthwise causal conv (width 4) + bias + SiLU on the x half of xz.
// ---------------------------------------------------------------------------
__global__ void __launch_bounds__(256) mamba_conv_silu(
    const float* __restrict__ xz, const float* __restrict__ cw,
    const float* __restrict__ cb, float* __restrict__ xin) {
  const int idx = blockIdx.x * 256 + threadIdx.x;
  const int d  = idx & (D_INNER - 1);
  const int bt = idx >> 11;                  // b*L_SEQ + t
  const int t  = bt & (L_SEQ - 1);
  float a = cb[d];
#pragma unroll
  for (int j = 0; j < D_CONV; ++j) {
    const int tt = t - (D_CONV - 1) + j;
    if (tt >= 0)
      a += cw[d * D_CONV + j] * xz[(size_t)(bt - (D_CONV - 1) + j) * N_XZ + d];
  }
  const float s = 1.0f / (1.0f + __expf(-a));
  xin[idx] = a * s;
}

// ---------------------------------------------------------------------------
// Selective scan: one thread per (b,d) channel, h[16] in registers,
// B/C staged in LDS in 64-timestep chunks. softplus(dt + bias) fused.
// ---------------------------------------------------------------------------
__global__ void __launch_bounds__(256) mamba_scan(
    const float* __restrict__ dtraw, const float* __restrict__ dtb,
    const float* __restrict__ A_log, const float* __restrict__ Dp,
    const float* __restrict__ dbc, const float* __restrict__ xin,
    float* __restrict__ y) {
  const int tid = threadIdx.x;
  const int d = blockIdx.x * 256 + tid;
  const int b = blockIdx.y;

  float Ad[D_STATE], h[D_STATE];
#pragma unroll
  for (int s = 0; s < D_STATE; ++s) {
    Ad[s] = -__expf(A_log[d * D_STATE + s]);
    h[s] = 0.0f;
  }
  const float Dd = Dp[d];
  const float bias = dtb[d];

  __shared__ float sBC[64][2 * D_STATE];     // [t][B 0..15 | C 16..31]
  for (int tc = 0; tc < L_SEQ; tc += 64) {
    __syncthreads();
#pragma unroll
    for (int i = 0; i < 8; ++i) {
      const int li = tid + i * 256;
      const int tt = li >> 5, c = li & 31;
      sBC[tt][c] = dbc[(size_t)(b * L_SEQ + tc + tt) * N_DBC + D_INNER + c];
    }
    __syncthreads();
    for (int j = 0; j < 64; ++j) {
      const size_t row = (size_t)(b * L_SEQ + tc + j);
      const float draw  = dtraw[row * D_INNER + d] + bias;
      const float delta = (draw > 20.0f) ? draw : log1pf(__expf(draw));
      const float xv = xin[row * D_INNER + d];
      const float dx = delta * xv;
      float accy = Dd * xv;
#pragma unroll
      for (int s = 0; s < D_STATE; ++s) {
        const float dA = __expf(delta * Ad[s]);
        h[s] = dA * h[s] + dx * sBC[j][s];
        accy += h[s] * sBC[j][D_STATE + s];
      }
      y[row * D_INNER + d] = accy;
    }
  }
}

// ---------------------------------------------------------------------------
// y * silu(z)
// ---------------------------------------------------------------------------
__global__ void __launch_bounds__(256) mamba_gate(
    const float* __restrict__ y, const float* __restrict__ xz,
    float* __restrict__ out) {
  const int idx = blockIdx.x * 256 + threadIdx.x;
  const int d  = idx & (D_INNER - 1);
  const int bt = idx >> 11;
  const float z = xz[(size_t)bt * N_XZ + D_INNER + d];
  const float s = z / (1.0f + __expf(-z));
  out[idx] = y[idx] * s;
}

// ---------------------------------------------------------------------------
extern "C" void kernel_launch(void* const* d_in, const int* in_sizes, int n_in,
                              void* d_out, int out_size, void* d_ws, size_t ws_size,
                              hipStream_t stream) {
  (void)in_sizes; (void)n_in; (void)out_size; (void)ws_size;
  const float* x          = (const float*)d_in[0];
  const float* in_proj_w  = (const float*)d_in[1];
  const float* conv_w     = (const float*)d_in[2];
  const float* conv_b     = (const float*)d_in[3];
  const float* x_proj_w   = (const float*)d_in[4];
  const float* dt_proj_w  = (const float*)d_in[5];
  const float* dt_proj_b  = (const float*)d_in[6];
  const float* A_log      = (const float*)d_in[7];
  const float* Dp         = (const float*)d_in[8];
  const float* out_proj_w = (const float*)d_in[9];
  float* out = (float*)d_out;

  float* ws    = (float*)d_ws;
  float* xz    = ws;                                   // BL * 4096 fp32
  float* xin   = xz    + (size_t)BL * N_XZ;            // BL * 2048
  float* dbc   = xin   + (size_t)BL * D_INNER;         // BL * 2080
  float* dtraw = dbc   + (size_t)BL * N_DBC;           // BL * 2048
  float* ybuf  = dtraw + (size_t)BL * D_INNER;         // BL * 2048
  float* ygate = dtraw;                                // reuse after scan

  // bf16 weight copies (8-byte aligned region after fp32 workspace)
  unsigned short* w_in  = (unsigned short*)(ybuf + (size_t)BL * D_INNER);
  unsigned short* w_x   = w_in + (size_t)N_XZ * D_MODEL;     // 2080*2048
  unsigned short* w_dt  = w_x  + (size_t)N_DBC * D_INNER;    // 2048*2048
  unsigned short* w_out = w_dt + (size_t)D_INNER * D_INNER;  // 1024*2048

  const dim3 blk(256);

  // 0. weight fp32 -> bf16 (once per launch; ~30MB total)
  {
    const int n_in4  = (N_XZ * D_MODEL) / 4;
    const int n_x4   = (N_DBC * D_INNER) / 4;
    const int n_dt4  = (D_INNER * D_INNER) / 4;
    const int n_out4 = (D_MODEL * D_INNER) / 4;
    mamba_cvt_bf16<<<dim3((n_in4 + 255) / 256), blk, 0, stream>>>(in_proj_w, w_in, n_in4);
    mamba_cvt_bf16<<<dim3((n_x4 + 255) / 256), blk, 0, stream>>>(x_proj_w, w_x, n_x4);
    mamba_cvt_bf16<<<dim3((n_dt4 + 255) / 256), blk, 0, stream>>>(dt_proj_w, w_dt, n_dt4);
    mamba_cvt_bf16<<<dim3((n_out4 + 255) / 256), blk, 0, stream>>>(out_proj_w, w_out, n_out4);
  }

  // 1. in_proj: xz = x @ in_proj_w^T
  mamba_gemm_wmma_bf16<<<dim3(BL / 128, N_XZ / 128), blk, 0, stream>>>(
      x, w_in, xz, N_XZ, D_MODEL, D_MODEL, D_MODEL, N_XZ);
  // 2. depthwise conv + SiLU -> xin
  mamba_conv_silu<<<dim3((BL * D_INNER) / 256), blk, 0, stream>>>(
      xz, conv_w, conv_b, xin);
  // 3. x_proj: dbc = xin @ x_proj_w^T  (N=2080, TDM zero-fills OOB rows)
  mamba_gemm_wmma_bf16<<<dim3(BL / 128, (N_DBC + 127) / 128), blk, 0, stream>>>(
      xin, w_x, dbc, N_DBC, D_INNER, D_INNER, D_INNER, N_DBC);
  // 4. dt_proj: dtraw = dbc[:, :2048] @ dt_proj_w^T  (lda = 2080)
  mamba_gemm_wmma_bf16<<<dim3(BL / 128, D_INNER / 128), blk, 0, stream>>>(
      dbc, w_dt, dtraw, D_INNER, D_INNER, N_DBC, D_INNER, D_INNER);
  // 5. selective scan -> ybuf
  mamba_scan<<<dim3(D_INNER / 256, B_SZ), blk, 0, stream>>>(
      dtraw, dt_proj_b, A_log, Dp, dbc, xin, ybuf);
  // 6. gate: ygate = ybuf * silu(z)
  mamba_gate<<<dim3((BL * D_INNER) / 256), blk, 0, stream>>>(ybuf, xz, ygate);
  // 7. out_proj: out = ygate @ out_proj_w^T
  mamba_gemm_wmma_bf16<<<dim3(BL / 128, D_MODEL / 128), blk, 0, stream>>>(
      ygate, w_out, out, D_MODEL, D_INNER, D_INNER, D_INNER, D_MODEL);
}
